// Net_60327110640112
// MI455X (gfx1250) — matrix operand order repaired and checked by reading, not compiled, not run on hardware
//
#include <hip/hip_runtime.h>
#include <hip/hip_bf16.h>

typedef _Float16 h16;
typedef __attribute__((ext_vector_type(16))) _Float16 v16h;
typedef __attribute__((ext_vector_type(8)))  float    v8f;

#define T_STEPS 2048
#define BATCH   128
#define IN_F    88
#define HID     200
#define OUT_F   88
#define NTG     13      // n-tiles per gate (pad 200 -> 208)
#define NT      52      // 4 gates * NTG
#define KTH     7       // K tiles for H (pad 200 -> 224, tiles of 32)
#define KTI     3       // K tiles for IN (pad 88 -> 96)
#define MT      8       // batch tiles of 16
#define FR      512     // halves per 16x32 fragment (32 lanes * 16)

// ---- workspace byte offsets ----
#define OFF_BIAS0 0u
#define OFF_BIAS1 4096u
#define OFF_BFC   8192u
#define OFF_BIH0  16384u                     // 52*3*512*2  = 159744
#define OFF_BHH0  (OFF_BIH0 + 159744u)       // 52*7*512*2  = 372736
#define OFF_BIH1  (OFF_BHH0 + 372736u)
#define OFF_BHH1  (OFF_BIH1 + 372736u)
#define OFF_BFCW  (OFF_BHH1 + 372736u)       // 6*7*512*2   = 43008
#define OFF_XP    (2u*1024u*1024u)           // 2048*8*3*512*2 = 50331648
#define OFF_Y0    (OFF_XP + 50331648u)       // 2048*8*7*512*2 = 117440512

__device__ __forceinline__ float fsig(float x)  { return 1.f / (1.f + __expf(-x)); }
__device__ __forceinline__ float ftanh(float x) { return 1.f - 2.f / (__expf(2.f * x) + 1.f); }

// ------------------------------------------------------------------
// Pack torch-layout weight w[N, K] into WMMA B-fragments (K x N tiles)
// B layout (16x16x32 f16): lane n-col = lane&15, K = (lane<16?0:8) + (e<8?e:e+8)
// gated: n-tiles grouped 13 per gate, real row = gate*200 + j
// ------------------------------------------------------------------
__global__ void pack_w(const float* __restrict__ w, h16* __restrict__ dst,
                       int ntiles, int ktiles, int Kreal, int gated, int Nreal) {
    long total = (long)ntiles * ktiles * FR;
    for (long idx = (long)blockIdx.x * blockDim.x + threadIdx.x; idx < total;
         idx += (long)gridDim.x * blockDim.x) {
        int e    = (int)(idx & 15);
        int lane = (int)((idx >> 4) & 31);
        long ft  = idx >> 9;
        int kt   = (int)(ft % ktiles);
        int nt   = (int)(ft / ktiles);
        int k    = kt * 32 + ((lane & 16) ? 8 : 0) + ((e < 8) ? e : e + 8);    // B-matrix K index
        int n, valid;
        if (gated) {
            int G = nt / NTG, jt = nt % NTG;
            int j = jt * 16 + (lane & 15);
            n = G * HID + j;
            valid = (j < HID) && (k < Kreal);
        } else {
            n = nt * 16 + (lane & 15);
            valid = (n < Nreal) && (k < Kreal);
        }
        dst[idx] = valid ? (h16)w[(long)n * Kreal + k] : (h16)0.f;
    }
}

// ------------------------------------------------------------------
// Pack x[T,B,88] f32 -> A-fragments f16 [t][mt][kt(3)][lane][e]
// A layout (16x32 f16): row m = lane&15, K = (lane<16?0:8) + (e<8?e:e+8)
// ------------------------------------------------------------------
__global__ void pack_x(const float* __restrict__ x, h16* __restrict__ dst) {
    long total = (long)T_STEPS * MT * KTI * FR;
    for (long idx = (long)blockIdx.x * blockDim.x + threadIdx.x; idx < total;
         idx += (long)gridDim.x * blockDim.x) {
        int e    = (int)(idx & 15);
        int lane = (int)((idx >> 4) & 31);
        long ft  = idx >> 9;
        int kt   = (int)(ft % KTI);
        long tm  = ft / KTI;
        int mt   = (int)(tm % MT);
        long t   = tm / MT;
        int k    = kt * 32 + ((lane & 16) ? 8 : 0) + ((e < 8) ? e : e + 8);
        int b    = mt * 16 + (lane & 15);
        dst[idx] = (k < IN_F) ? (h16)x[(t * BATCH + b) * IN_F + k] : (h16)0.f;
    }
}

__global__ void pack_bias(const float* __restrict__ bi0, const float* __restrict__ bh0,
                          const float* __restrict__ bi1, const float* __restrict__ bh1,
                          const float* __restrict__ bfc,
                          float* __restrict__ d0, float* __restrict__ d1,
                          float* __restrict__ dfc) {
    int i = blockIdx.x * blockDim.x + threadIdx.x;
    if (i < 4 * HID) { d0[i] = bi0[i] + bh0[i]; d1[i] = bi1[i] + bh1[i]; }
    if (i < 96) dfc[i] = (i < OUT_F) ? bfc[i] : 0.f;
}

// ------------------------------------------------------------------
// Persistent per-batch-tile LSTM layer. 13 waves (416 thr); wave = j-tile.
// Each wave owns the 4 gate output tiles (i,f,g,o) for its 16 hidden units.
// h state lives in LDS directly in A-fragment layout. Input-path weights
// are cached in LDS (all of them for L0; 5/7 k-tiles + FC weights for L1);
// recurrent weights stream from L2 each step. Per gate we issue ALL global
// weight loads first, run LDS-fed wmmas under them, then retire the global
// fragments into wmmas — deep clause + partial loadcnt waits instead of a
// depth-1 load/wait/wmma lockstep.
// ------------------------------------------------------------------
template <int LAYER>
__global__ __launch_bounds__(416) void lstm_kernel(
    const h16* __restrict__ inFrag,          // Xpack (L0) or Y0pack (L1)
    const h16* __restrict__ Bih, const h16* __restrict__ Bhh,
    const float* __restrict__ bias,
    h16* __restrict__ y0out,                 // L0 only
    const h16* __restrict__ Bfc, const float* __restrict__ bfc,
    float* __restrict__ out,                 // L1 only
    float* __restrict__ hn, float* __restrict__ cn) {
    constexpr int IKT  = (LAYER == 0) ? KTI : KTH;   // input k-tiles
    constexpr int IKTL = (LAYER == 0) ? KTI : 5;     // k-tiles cached in LDS
    constexpr int IKTG = IKT - IKTL;                 // input k-tiles streamed
    constexpr int FCSZ = (LAYER == 1) ? 6 * KTH * FR : 32;

    __shared__ __align__(32) h16 hfrag[KTH * FR];        //   7.0 KB
    __shared__ __align__(32) h16 ihlds[NT * IKTL * FR];  // 156 / 260 KB
    __shared__ __align__(32) h16 fclds[FCSZ];            //   - / 42 KB

    const int tid  = threadIdx.x;
    const int wave = tid >> 5;
    const int lane = tid & 31;
    const int mt   = blockIdx.x;

    // cooperative fill: ih weight fragments -> LDS (subset of k-tiles)
    for (int c = tid; c < NT * IKTL * 32; c += blockDim.x) {
        int frag = c >> 5, l = c & 31;
        int nt = frag / IKTL, kt = frag % IKTL;
        *(v16h*)&ihlds[(long)frag * FR + l * 16] =
            *(const v16h*)(Bih + ((long)nt * IKT + kt) * FR + l * 16);
    }
    if (LAYER == 1) {
        for (int c = tid; c < 6 * KTH * 32; c += blockDim.x)
            *(v16h*)&fclds[(long)c * 16] = *(const v16h*)(Bfc + (long)c * 16);
    }
    for (int i = tid; i < KTH * FR; i += blockDim.x) hfrag[i] = (h16)0.f;
    __syncthreads();

    const int  col    = lane & 15;
    const int  jt     = wave;                 // 0..12
    const int  j      = jt * 16 + col;        // hidden unit (may pad >= 200)
    const bool jvalid = (j < HID);
    const int  m0     = (lane < 16) ? 0 : 8;  // C/D row base for this lane

    float bg[4];
#pragma unroll
    for (int G = 0; G < 4; ++G) bg[G] = jvalid ? bias[G * HID + j] : 0.f;

    // scatter mapping: h value (m, j) -> A-fragment slot in LDS
    const int k32  = j & 31;
    const int skt  = j >> 5;
    const int half = (k32 >> 3) & 1;
    const int se   = (k32 & 7) + ((k32 & 16) ? 8 : 0);
    h16* hstore = &hfrag[skt * FR + (half * 16 + m0) * 16 + se];

    const int ocol = wave * 16 + col;        // FC output column (L1)
    float bfcv = 0.f;
    if (LAYER == 1 && wave < 6) bfcv = (ocol < OUT_F) ? bfc[ocol] : 0.f;

    v8f cv = {};
    v8f hv = {};

    for (int t = 0; t < T_STEPS; ++t) {
        // Anti-LICM: unknown-but-zero SGPR offset makes the streamed-weight
        // addresses loop-variant (no hoist -> no spill) while `Base + woff`
        // keeps global address-space provenance (global_load, not flat_load).
        long woff = 0;
        asm volatile("" : "+s"(woff));
        const h16* bhh_t = Bhh + woff;
        const h16* bih_t = Bih + woff;

        v8f acc[4];
#pragma unroll
        for (int G = 0; G < 4; ++G) {
            v8f a;
#pragma unroll
            for (int r = 0; r < 8; ++r) a[r] = bg[G];
            acc[G] = a;
        }
        // h_{t-1} A-fragments (contiguous 32B LDS reads)
        v16h hf[KTH];
#pragma unroll
        for (int kt = 0; kt < KTH; ++kt)
            hf[kt] = *(const v16h*)&hfrag[kt * FR + lane * 16];
        // input A-fragments + prefetch of next step's
        const h16* inb = inFrag + ((long)t * MT + mt) * IKT * FR + lane * 16;
        v16h xf[IKT];
#pragma unroll
        for (int kt = 0; kt < IKT; ++kt) xf[kt] = *(const v16h*)(inb + (long)kt * FR);
        if (t + 1 < T_STEPS)
            __builtin_prefetch(inFrag + ((long)(t + 1) * MT + mt) * IKT * FR + lane * 16, 0, 1);

#pragma unroll
        for (int G = 0; G < 4; ++G) {
            // 1) issue ALL streamed weight loads for this gate (deep clause)
            v16h whh[KTH];
#pragma unroll
            for (int kt = 0; kt < KTH; ++kt)
                whh[kt] = *(const v16h*)(bhh_t + ((long)(G * NTG + jt) * KTH + kt) * FR + lane * 16);
            v16h wihg[IKTG > 0 ? IKTG : 1];
#pragma unroll
            for (int kt = 0; kt < IKTG; ++kt)
                wihg[kt] = *(const v16h*)(bih_t + ((long)(G * NTG + jt) * IKT + IKTL + kt) * FR + lane * 16);

            // 2) LDS-fed wmmas execute while the global loads are in flight
#pragma unroll
            for (int kt = 0; kt < IKTL; ++kt) {
                v16h bf = *(const v16h*)&ihlds[((long)(G * NTG + jt) * IKTL + kt) * FR + lane * 16];
                acc[G] = __builtin_amdgcn_wmma_f32_16x16x32_f16(
                    false, xf[kt], false, bf, (short)0, acc[G], false, false);
            }
            // 3) retire the global fragments into wmmas (partial waits)
#pragma unroll
            for (int kt = 0; kt < IKTG; ++kt)
                acc[G] = __builtin_amdgcn_wmma_f32_16x16x32_f16(
                    false, xf[IKTL + kt], false, wihg[kt], (short)0, acc[G], false, false);
#pragma unroll
            for (int kt = 0; kt < KTH; ++kt)
                acc[G] = __builtin_amdgcn_wmma_f32_16x16x32_f16(
                    false, hf[kt], false, whh[kt], (short)0, acc[G], false, false);
        }
        __syncthreads();   // all waves done reading h_{t-1}

        // gate nonlinearities + state update (rows m0..m0+7, column j)
#pragma unroll
        for (int r = 0; r < 8; ++r) {
            float ig = fsig(acc[0][r]);
            float fg = fsig(acc[1][r]);
            float gg = ftanh(acc[2][r]);
            float og = fsig(acc[3][r]);
            float cn_ = fg * cv[r] + ig * gg;
            cv[r] = cn_;
            float h_ = og * ftanh(cn_);
            hv[r] = jvalid ? h_ : 0.f;
            hstore[r * 16] = (h16)hv[r];     // write h_t as A-fragment
        }
        __syncthreads();   // h_t visible to everyone

        if constexpr (LAYER == 0) {
            if (wave < KTH) {                // stream y0 A-fragments to global
                v16h f = *(const v16h*)&hfrag[wave * FR + lane * 16];
                *(v16h*)(y0out + (((long)t * MT + mt) * KTH + wave) * FR + lane * 16) = f;
            }
        } else {
            if (wave < 6) {                  // fused FC + sigmoid epilogue
                v8f oa;
#pragma unroll
                for (int r = 0; r < 8; ++r) oa[r] = bfcv;
#pragma unroll
                for (int kt = 0; kt < KTH; ++kt) {
                    v16h af = *(const v16h*)&hfrag[kt * FR + lane * 16];
                    v16h bf = *(const v16h*)&fclds[(long)(wave * KTH + kt) * FR + lane * 16];
                    oa = __builtin_amdgcn_wmma_f32_16x16x32_f16(
                        false, af, false, bf, (short)0, oa, false, false);
                }
                if (ocol < OUT_F) {
#pragma unroll
                    for (int r = 0; r < 8; ++r) {
                        int b = mt * 16 + m0 + r;
                        out[((long)t * BATCH + b) * OUT_F + ocol] = fsig(oa[r]);
                    }
                }
            }
        }
    }

    if (jvalid) {                            // final hidden / cell state
#pragma unroll
        for (int r = 0; r < 8; ++r) {
            int b = mt * 16 + m0 + r;
            hn[(long)LAYER * BATCH * HID + (long)b * HID + j] = hv[r];
            cn[(long)LAYER * BATCH * HID + (long)b * HID + j] = cv[r];
        }
    }
}

extern "C" void kernel_launch(void* const* d_in, const int* in_sizes, int n_in,
                              void* d_out, int out_size, void* d_ws, size_t ws_size,
                              hipStream_t stream) {
    (void)in_sizes; (void)n_in; (void)out_size; (void)ws_size;
    const float* x     = (const float*)d_in[0];
    const float* w_ih0 = (const float*)d_in[1];
    const float* w_hh0 = (const float*)d_in[2];
    const float* b_ih0 = (const float*)d_in[3];
    const float* b_hh0 = (const float*)d_in[4];
    const float* w_ih1 = (const float*)d_in[5];
    const float* w_hh1 = (const float*)d_in[6];
    const float* b_ih1 = (const float*)d_in[7];
    const float* b_hh1 = (const float*)d_in[8];
    const float* w_fc  = (const float*)d_in[9];
    const float* b_fc  = (const float*)d_in[10];

    char* ws = (char*)d_ws;
    float* bias0 = (float*)(ws + OFF_BIAS0);
    float* bias1 = (float*)(ws + OFF_BIAS1);
    float* bfc   = (float*)(ws + OFF_BFC);
    h16* Bih0 = (h16*)(ws + OFF_BIH0);
    h16* Bhh0 = (h16*)(ws + OFF_BHH0);
    h16* Bih1 = (h16*)(ws + OFF_BIH1);
    h16* Bhh1 = (h16*)(ws + OFF_BHH1);
    h16* Bfcw = (h16*)(ws + OFF_BFCW);
    h16* Xp   = (h16*)(ws + OFF_XP);
    h16* Y0p  = (h16*)(ws + OFF_Y0);

    float* out = (float*)d_out;
    float* hn  = out + (long)T_STEPS * BATCH * OUT_F;
    float* cn  = hn + 2 * BATCH * HID;

    pack_bias<<<4, 256, 0, stream>>>(b_ih0, b_hh0, b_ih1, b_hh1, b_fc, bias0, bias1, bfc);
    pack_w<<<312, 256, 0, stream>>>(w_ih0, Bih0, NT, KTI, IN_F, 1, 0);
    pack_w<<<728, 256, 0, stream>>>(w_hh0, Bhh0, NT, KTH, HID, 1, 0);
    pack_w<<<728, 256, 0, stream>>>(w_ih1, Bih1, NT, KTH, HID, 1, 0);
    pack_w<<<728, 256, 0, stream>>>(w_hh1, Bhh1, NT, KTH, HID, 1, 0);
    pack_w<<<84, 256, 0, stream>>>(w_fc, Bfcw, 6, KTH, HID, 0, OUT_F);
    pack_x<<<8192, 256, 0, stream>>>(x, Xp);

    lstm_kernel<0><<<MT, 416, 0, stream>>>(Xp, Bih0, Bhh0, bias0, Y0p,
                                           nullptr, nullptr, nullptr, hn, cn);
    lstm_kernel<1><<<MT, 416, 0, stream>>>(Y0p, Bih1, Bhh1, bias1, nullptr,
                                           Bfcw, bfc, out, hn, cn);
}
